// CAP_11218454577547
// MI455X (gfx1250) — compile-verified, matrix operand order checked by reading, not run on hardware
//
#include <hip/hip_runtime.h>
#include <hip/hip_bf16.h>

#define DIM     768
#define SEQ     256
#define KCAND   8
#define BATCH   32
#define PLEN    16
#define NSTREAM 2
#define SCALE   0.03608439182435161f   // 1/sqrt(768)

// LDS layout for the attention kernel (bytes)
#define SROW    260                    // f32 score row stride (pad: conflict-free WMMA stores)
#define RROW    776                    // f16 staged-r row stride
#define PPROW   264                    // f16 pooled-prob row stride
#define RTROW   40                     // f16 transposed-r row stride (32 + pad)
#define LDS_S_OFF  0
#define LDS_R_OFF  (SEQ * SROW * 4)                 // 266240
#define LDS_PP_OFF (LDS_R_OFF + 32 * RROW * 2)      // 315904
#define LDS_TOTAL  (LDS_PP_OFF + 16 * PPROW * 2)    // 324352 <= 320KB+rounding

typedef __attribute__((ext_vector_type(16))) _Float16 v16h;
typedef __attribute__((ext_vector_type(8)))  _Float16 v8h;
typedef __attribute__((ext_vector_type(4)))  _Float16 v4h;
typedef __attribute__((ext_vector_type(8)))  float    v8f;
typedef __attribute__((ext_vector_type(4)))  float    v4f;

union V16H { v16h v; v8h h[2]; _Float16 e[16]; };

__device__ __forceinline__ v8f wmma_f16(v16h a, v16h b, v8f c) {
  return __builtin_amdgcn_wmma_f32_16x16x32_f16(false, a, false, b, (short)0, c,
                                                false, false);
}

// ---------------------------------------------------------------------------
// k_prep: Mt[g][f] = scale * sum_e Wq[e][f]*Wk[e][g]  (f16, stored transposed)
//         u[g]     = scale * sum_e Wk[e][g]*bq[e]
// ---------------------------------------------------------------------------
__global__ __launch_bounds__(256) void k_prep(const float* __restrict__ Wq,
                                              const float* __restrict__ Wk,
                                              const float* __restrict__ bq,
                                              _Float16* __restrict__ Mt,
                                              float* __restrict__ u) {
  int idx = blockIdx.x * 256 + threadIdx.x;   // [0, 768*768)
  int g = idx / DIM, f = idx % DIM;
  float acc = 0.f;
  for (int e = 0; e < DIM; ++e) acc += Wq[e * DIM + f] * Wk[e * DIM + g];
  Mt[g * DIM + f] = (_Float16)(acc * SCALE);
  if (idx < DIM) {
    float a = 0.f;
    for (int e = 0; e < DIM; ++e) a += Wk[e * DIM + idx] * bq[e];
    u[idx] = a * SCALE;
  }
}

__global__ __launch_bounds__(256) void k_zero(float* __restrict__ p, int n) {
  int i = blockIdx.x * 256 + threadIdx.x;
  if (i < n) p[i] = 0.f;
}

// ---------------------------------------------------------------------------
// k_gproj: G[16384][768] f16 = x @ Mt^T + u ; rows 0..8191 = T, 8192.. = V
// one (16x16) output tile per wave, K=768 in 24 WMMA steps
// ---------------------------------------------------------------------------
__global__ __launch_bounds__(256) void k_gproj(const float* __restrict__ T,
                                               const float* __restrict__ V,
                                               const _Float16* __restrict__ Mt,
                                               const float* __restrict__ u,
                                               _Float16* __restrict__ G) {
  int lane = threadIdx.x & 31, wave = threadIdx.x >> 5;
  int half = lane >> 4, l15 = lane & 15;
  int tile = blockIdx.x * 8 + wave;           // 1024*48 tiles
  int mt = tile / 48, nt = tile % 48;
  int R0 = mt * 16;
  const float* X = (R0 < BATCH * SEQ) ? T : V;        // stream 0 = T
  int rl = R0 & (BATCH * SEQ - 1);
  const float* arow = X + (size_t)(rl + l15) * DIM;
  int g = nt * 16 + l15;
  const _Float16* brow = Mt + (size_t)g * DIM;
  v8f acc = {};
  for (int d = 0; d < 24; ++d) {
    int k0 = d * 32 + half * 8;
    V16H A, B;
    v4f a0 = *(const v4f*)(arow + k0);
    v4f a1 = *(const v4f*)(arow + k0 + 4);
    v4f a2 = *(const v4f*)(arow + k0 + 16);
    v4f a3 = *(const v4f*)(arow + k0 + 20);
#pragma unroll
    for (int j = 0; j < 4; ++j) {
      A.e[j] = (_Float16)a0[j];      A.e[4 + j] = (_Float16)a1[j];
      A.e[8 + j] = (_Float16)a2[j];  A.e[12 + j] = (_Float16)a3[j];
    }
    B.h[0] = *(const v8h*)(brow + k0);
    B.h[1] = *(const v8h*)(brow + k0 + 16);
    acc = wmma_f16(A.v, B.v, acc);
  }
  float ub = u[g];
#pragma unroll
  for (int i = 0; i < 8; ++i) {
    int row = R0 + i + half * 8;
    G[(size_t)row * DIM + g] = (_Float16)(acc[i] + ub);
  }
}

// ---------------------------------------------------------------------------
// k_attn: one workgroup per (stream, b, kc).
//   pass1: S[256][256] = G_b . r^T  (f16 WMMA, f32 scores in LDS)
//   softmax rows, fold pool(1/16)*Kmean(1/8) -> Pp[16][256] f16
//   pass2: Y[16][768] = Pp . r ; atomicAdd into Acc[stream][b]
// ---------------------------------------------------------------------------
__global__ __launch_bounds__(256) void k_attn(const float* __restrict__ r_t,
                                              const float* __restrict__ r_i,
                                              const _Float16* __restrict__ G,
                                              float* __restrict__ Acc) {
  extern __shared__ char smem[];
  float* Sl = (float*)(smem + LDS_S_OFF);          // [256][SROW] f32
  _Float16* Rl = (_Float16*)(smem + LDS_R_OFF);    // [32][RROW] f16
  _Float16* Pp = (_Float16*)(smem + LDS_PP_OFF);   // [16][PPROW] f16
  _Float16* RT = (_Float16*)(smem);                // pass2: [768][RTROW] f16

  int tid = threadIdx.x;
  int lane = tid & 31, wave = tid >> 5, half = lane >> 4, l15 = lane & 15;
  int wg = blockIdx.x;
  int st = wg >> 8;                  // 0 = T-stream, 1 = V-stream
  int b = (wg >> 3) & 31;
  int kc = wg & 7;
  const float* Rg = (st == 0 ? r_t : r_i) + (size_t)(b * KCAND + kc) * SEQ * DIM;
  const _Float16* Gs = G + (size_t)(st * BATCH + b) * SEQ * DIM;
  float* AccS = Acc + (size_t)(st * BATCH + b) * PLEN * DIM;

  // ---------------- pass 1: scores ----------------
  for (int sc = 0; sc < 8; ++sc) {
    // stage 32 r-rows as f16, row-major
    for (int it = 0; it < 24; ++it) {
      int idx = it * 256 + tid;                    // < 6144
      int srow = idx / 192, c4 = (idx % 192) * 4;
      v4f v = *(const v4f*)(Rg + (size_t)(sc * 32 + srow) * DIM + c4);
      v4h h = { (_Float16)v[0], (_Float16)v[1], (_Float16)v[2], (_Float16)v[3] };
      *(v4h*)(Rl + srow * RROW + c4) = h;
    }
    __syncthreads();
    if (sc + 1 < 8 && tid < 32)
      __builtin_prefetch(Rg + (size_t)((sc + 1) * 32 + tid) * DIM, 0, 0);

    for (int m = wave; m < 16; m += 8) {
      v8f acc0 = {}, acc1 = {};
      const _Float16* ga = Gs + (size_t)(m * 16 + l15) * DIM;
      for (int d = 0; d < 24; ++d) {
        int k0 = d * 32 + half * 8;
        V16H A, B0, B1;
        A.h[0] = *(const v8h*)(ga + k0);
        A.h[1] = *(const v8h*)(ga + k0 + 16);
        B0.h[0] = *(const v8h*)(Rl + l15 * RROW + k0);
        B0.h[1] = *(const v8h*)(Rl + l15 * RROW + k0 + 16);
        B1.h[0] = *(const v8h*)(Rl + (16 + l15) * RROW + k0);
        B1.h[1] = *(const v8h*)(Rl + (16 + l15) * RROW + k0 + 16);
        acc0 = wmma_f16(A.v, B0.v, acc0);
        acc1 = wmma_f16(A.v, B1.v, acc1);
      }
#pragma unroll
      for (int i = 0; i < 8; ++i) {
        int row = m * 16 + i + half * 8;
        Sl[row * SROW + sc * 32 + l15] = acc0[i];
        Sl[row * SROW + sc * 32 + 16 + l15] = acc1[i];
      }
    }
    __syncthreads();
  }

  // ---------------- softmax: one row per thread ----------------
  {
    float* rp = Sl + tid * SROW;
    float mx = rp[0];
    for (int i = 1; i < SEQ; ++i) mx = fmaxf(mx, rp[i]);
    float sum = 0.f;
    for (int i = 0; i < SEQ; ++i) { float e = __expf(rp[i] - mx); rp[i] = e; sum += e; }
    float inv = 1.f / sum;
    for (int i = 0; i < SEQ; ++i) rp[i] *= inv;
  }
  __syncthreads();

  // ---------------- pooled probs: Pp = (1/128) * pool(P) ----------------
  {
    int p = tid >> 4;
    int c0 = (tid & 15) * 16;
    for (int c = c0; c < c0 + 16; ++c) {
      float s = 0.f;
      for (int rr = 0; rr < 16; ++rr) s += Sl[(p * 16 + rr) * SROW + c];
      Pp[p * PPROW + c] = (_Float16)(s * (1.f / 128.f));
    }
  }
  __syncthreads();

  // ---------------- pass 2: Y = Pp @ r ----------------
  v8f zero = {};
  v8f acc[6];
#pragma unroll
  for (int j = 0; j < 6; ++j) acc[j] = zero;
  for (int sc = 0; sc < 8; ++sc) {
    // stage 32 r-rows transposed: RT[d][s]
    for (int it = 0; it < 24; ++it) {
      int idx = it * 256 + tid;
      int srow = idx / 192, c4 = (idx % 192) * 4;
      v4f v = *(const v4f*)(Rg + (size_t)(sc * 32 + srow) * DIM + c4);
#pragma unroll
      for (int j = 0; j < 4; ++j) RT[(c4 + j) * RTROW + srow] = (_Float16)v[j];
    }
    __syncthreads();
    V16H A;
    A.h[0] = *(const v8h*)(Pp + l15 * PPROW + sc * 32 + half * 8);
    A.h[1] = *(const v8h*)(Pp + l15 * PPROW + sc * 32 + half * 8 + 16);
#pragma unroll
    for (int j = 0; j < 6; ++j) {
      int nt = wave * 6 + j;
      V16H B;
      B.h[0] = *(const v8h*)(RT + (nt * 16 + l15) * RTROW + half * 8);
      B.h[1] = *(const v8h*)(RT + (nt * 16 + l15) * RTROW + 16 + half * 8);
      acc[j] = wmma_f16(A.v, B.v, acc[j]);
    }
    __syncthreads();
  }
#pragma unroll
  for (int j = 0; j < 6; ++j) {
    int nt = wave * 6 + j;
#pragma unroll
    for (int i = 0; i < 8; ++i) {
      int p = i + half * 8;
      atomicAdd(&AccS[p * DIM + nt * 16 + l15], acc[j][i]);
    }
  }
}

// ---------------------------------------------------------------------------
// k_out: out[1024][768] = Acc @ Wv^T + bv  (rows already in tuple order: T,V)
// ---------------------------------------------------------------------------
__global__ __launch_bounds__(256) void k_out(const float* __restrict__ Acc,
                                             const float* __restrict__ Wv,
                                             const float* __restrict__ bv,
                                             float* __restrict__ out) {
  int lane = threadIdx.x & 31, wave = threadIdx.x >> 5;
  int half = lane >> 4, l15 = lane & 15;
  int tile = blockIdx.x * 8 + wave;            // 64*48 tiles
  int mt = tile / 48, nt = tile % 48;
  const float* arow = Acc + (size_t)(mt * 16 + l15) * DIM;
  const float* brow = Wv + (size_t)(nt * 16 + l15) * DIM;   // B[k=f][n=e]=Wv[e][f]
  v8f acc = {};
  for (int d = 0; d < 24; ++d) {
    int k0 = d * 32 + half * 8;
    V16H A, B;
    v4f a0 = *(const v4f*)(arow + k0);
    v4f a1 = *(const v4f*)(arow + k0 + 4);
    v4f a2 = *(const v4f*)(arow + k0 + 16);
    v4f a3 = *(const v4f*)(arow + k0 + 20);
    v4f b0 = *(const v4f*)(brow + k0);
    v4f b1 = *(const v4f*)(brow + k0 + 4);
    v4f b2 = *(const v4f*)(brow + k0 + 16);
    v4f b3 = *(const v4f*)(brow + k0 + 20);
#pragma unroll
    for (int j = 0; j < 4; ++j) {
      A.e[j] = (_Float16)a0[j];      A.e[4 + j] = (_Float16)a1[j];
      A.e[8 + j] = (_Float16)a2[j];  A.e[12 + j] = (_Float16)a3[j];
      B.e[j] = (_Float16)b0[j];      B.e[4 + j] = (_Float16)b1[j];
      B.e[8 + j] = (_Float16)b2[j];  B.e[12 + j] = (_Float16)b3[j];
    }
    acc = wmma_f16(A.v, B.v, acc);
  }
  float bb = bv[nt * 16 + l15];
#pragma unroll
  for (int i = 0; i < 8; ++i) {
    int row = mt * 16 + i + half * 8;
    out[(size_t)row * DIM + nt * 16 + l15] = acc[i] + bb;
  }
}

// ---------------------------------------------------------------------------
// workspace layout (bytes):
//   [0, 1179648)          Mt  f16 [768][768]
//   [1179648, 1182720)    u   f32 [768] (+pad)
//   [1182720, 4328448)    Acc f32 [2][32][16][768]
//   [4328448, 29494272)   G   f16 [2][32][256][768]
// ---------------------------------------------------------------------------
extern "C" void kernel_launch(void* const* d_in, const int* in_sizes, int n_in,
                              void* d_out, int out_size, void* d_ws, size_t ws_size,
                              hipStream_t stream) {
  const float* V   = (const float*)d_in[0];
  const float* T   = (const float*)d_in[1];
  const float* r_i = (const float*)d_in[2];
  const float* r_t = (const float*)d_in[3];
  const float* Wq  = (const float*)d_in[4];
  const float* bq  = (const float*)d_in[5];
  const float* Wk  = (const float*)d_in[6];
  // d_in[7] = bk: provably cancels in softmax (constant along s) -> unused
  const float* Wv  = (const float*)d_in[8];
  const float* bv  = (const float*)d_in[9];
  float* out = (float*)d_out;

  char* ws = (char*)d_ws;
  _Float16* Mt = (_Float16*)(ws + 0);
  float*    u  = (float*)(ws + 1179648);
  float*    Acc = (float*)(ws + 1182720);
  _Float16* G  = (_Float16*)(ws + 4328448);

  (void)in_sizes; (void)n_in; (void)out_size; (void)ws_size;
  hipFuncSetAttribute(reinterpret_cast<const void*>(k_attn),
                      hipFuncAttributeMaxDynamicSharedMemorySize, LDS_TOTAL);

  k_prep<<<(DIM * DIM) / 256, 256, 0, stream>>>(Wq, Wk, bq, Mt, u);
  k_zero<<<(NSTREAM * BATCH * PLEN * DIM) / 256, 256, 0, stream>>>(
      Acc, NSTREAM * BATCH * PLEN * DIM);
  k_gproj<<<(NSTREAM * BATCH * SEQ / 16) * 48 / 8, 256, 0, stream>>>(T, V, Mt, u, G);
  k_attn<<<NSTREAM * BATCH * KCAND, 256, LDS_TOTAL, stream>>>(r_t, r_i, G, Acc);
  k_out<<<(NSTREAM * BATCH * PLEN / 16) * 48 / 8, 256, 0, stream>>>(Acc, Wv, bv, out);
}
